// SprecherLayerBlock_44598940401668
// MI455X (gfx1250) — matrix operand order repaired and checked by reading, not compiled, not run on hardware
//
#include <hip/hip_runtime.h>

typedef __attribute__((ext_vector_type(16))) _Float16 v16h;
typedef __attribute__((ext_vector_type(8)))  float    v8f;

#define B_SZ 1024
#define P_SZ 256
#define Q_SZ 512
#define K_SZ 64

#define PHI_MIN (-6.0f)
#define PHI_MAX (7.0f)

// CDNA5 async global->LDS copy path (guarded; falls back to plain LDS staging).
// The clang builtin takes int-pointee address-space pointers.
#if defined(__AMDGCN__) && __has_builtin(__builtin_amdgcn_global_load_async_to_lds_b32)
#define HAVE_ASYNC_LDS 1
typedef int __attribute__((address_space(1)))* as1_ip;
typedef int __attribute__((address_space(3)))* as3_ip;
// generic->AS1: numeric value is the global address; generic->AS3: low 32 bits
// of a generic LDS pointer are the LDS offset (ISA 10.2 aperture mapping).
#define AS1(p) ((as1_ip)(unsigned long long)(const void*)(p))
#define AS3(p) ((as3_ip)(unsigned int)(unsigned long long)(const void*)(p))
#endif

// ---------------------------------------------------------------------------
// Kernel 1: build spline LUTs in d_ws.
//   ws[0..127]   : phi table, float2 {c[i], c[i+1]-c[i]} x 64 (monotonic coeffs)
//   ws[128..255] : Phi table, float2 {C[i], C[i+1]-C[i]} x 64
// K=64 softplus->cumsum->normalize is inherently serial and tiny: one lane.
// ---------------------------------------------------------------------------
__global__ void build_tables_kernel(const float* __restrict__ phi_log_inc,
                                    const float* __restrict__ Phi_coeffs,
                                    float* __restrict__ ws) {
  if (threadIdx.x == 0) {
    float c[K_SZ];
    float acc = 0.0f;
    for (int i = 0; i < K_SZ; ++i) {
      float v = phi_log_inc[i];
      float sp = (v > 20.0f) ? v : log1pf(expf(v));  // softplus
      acc += sp;
      c[i] = acc;                                     // cumsum
    }
    float inv = 1.0f / (acc + 1e-8f);                 // normalize to [0,1]
    for (int i = 0; i < K_SZ; ++i) c[i] *= inv;
    for (int i = 0; i < K_SZ - 1; ++i) {
      ws[2 * i]     = c[i];
      ws[2 * i + 1] = c[i + 1] - c[i];
    }
    ws[2 * (K_SZ - 1)]     = c[K_SZ - 1];
    ws[2 * (K_SZ - 1) + 1] = 0.0f;
    for (int i = 0; i < K_SZ - 1; ++i) {
      ws[128 + 2 * i]     = Phi_coeffs[i];
      ws[128 + 2 * i + 1] = Phi_coeffs[i + 1] - Phi_coeffs[i];
    }
    ws[128 + 2 * (K_SZ - 1)]     = Phi_coeffs[K_SZ - 1];
    ws[128 + 2 * (K_SZ - 1) + 1] = 0.0f;
  }
}

// ---------------------------------------------------------------------------
// Kernel 2: one block per batch row b. 8 waves; each wave owns 4 q-tiles of 16.
// Per q-tile: 8 chained v_wmma_f32_16x16x32_f16 reduce over P=256.
//   A[m=q, k=p]  = f16 phi(x[b,p] + shift[q])   (built on the fly, LDS LUT)
//   B[k=p, n]    = f16 lambda[p] replicated over all 16 columns (hoisted)
//   D[m, n]      = s[b, qbase+m] (dup over n); lanes 0/16 apply Phi + store.
// x row is pre-scaled into spline u-space at staging so each eval is just
// add -> floor/med3 -> LDS float2 gather -> fma -> cvt f16.
// ---------------------------------------------------------------------------
__launch_bounds__(256)
__global__ void sprecher_kernel(const float* __restrict__ x,
                                const float* __restrict__ lambdas,
                                const float* __restrict__ tabs,
                                float* __restrict__ out) {
  __shared__ __align__(16) float    xs[P_SZ];        // pre-scaled x row (u-space)
  __shared__ __align__(8)  float    phi_tab[2 * K_SZ];
  __shared__ __align__(8)  float    Phi_tab[2 * K_SZ];
  __shared__ __align__(4)  _Float16 lam_h[P_SZ];

  const int tid = threadIdx.x;
  const int b   = blockIdx.x;

  const float inv_h = 63.0f / (PHI_MAX - PHI_MIN);
  const float bias  = -PHI_MIN * inv_h;
  const float eta_s = (1.0f / (float)Q_SZ) * inv_h;  // shift step in u-space

  // --- staging ---
  xs[tid]    = fmaf(x[b * P_SZ + tid], inv_h, bias); // pre-scale to u-space
  lam_h[tid] = (_Float16)lambdas[tid];
#if defined(HAVE_ASYNC_LDS)
  {
    // Stage both LUTs through the CDNA5 async global->LDS DMA path.
    const float* src = tabs + tid;
    float*       dst = (tid < 128) ? &phi_tab[tid] : &Phi_tab[tid - 128];
    __builtin_amdgcn_global_load_async_to_lds_b32(AS1(src), AS3(dst), 0, 0);
  }
  asm volatile("s_wait_asynccnt 0x0" ::: "memory");
#else
  if (tid < 128) phi_tab[tid] = tabs[tid];
  else           Phi_tab[tid - 128] = tabs[tid];
#endif
  __syncthreads();

  const int lane = tid & 31;
  const int wave = tid >> 5;
  const int g    = lane >> 4;   // half-wave group (K-split per WMMA layout)
  const int m    = lane & 15;   // row within 16x16 tile = q offset

  const float2* ptab = (const float2*)phi_tab;
  const float2* Ptab = (const float2*)Phi_tab;

  // B fragments: lambda, identical in every column; invariant across q-tiles.
  // ISA 16-bit B layout: lane-half g, element j -> K = (j&7) + 8*g + 16*(j>>3).
  v16h bfr[8];
  #pragma unroll
  for (int ch = 0; ch < 8; ++ch) {
    #pragma unroll
    for (int j = 0; j < 16; ++j) {
      bfr[ch][j] = lam_h[ch * 32 + (j & 7) + g * 8 + ((j >> 3) << 4)];
    }
  }

  for (int qt = 0; qt < 4; ++qt) {
    const int   qbase   = (wave * 4 + qt) * 16;
    const float shift_s = (float)(qbase + m) * eta_s;  // u-space shift

    v8f acc = {0.f, 0.f, 0.f, 0.f, 0.f, 0.f, 0.f, 0.f};

    #pragma unroll
    for (int ch = 0; ch < 8; ++ch) {           // P chunks of 32
      // A fragment: 16 f16 spline values; ISA 16-bit A layout:
      // lane-half g, element j -> K = (j&7) + 8*g + 16*(j>>3)
      v16h afrag;
      #pragma unroll
      for (int j = 0; j < 16; ++j) {
        const int   k   = (j & 7) + g * 8 + ((j >> 3) << 4);
        const float u   = xs[ch * 32 + k] + shift_s;
        float       fi  = floorf(u);
        fi              = fminf(fmaxf(fi, 0.0f), 62.0f);
        const int   idx = (int)fi;
        const float t   = u - fi;              // unclamped: linear extrapolation
        const float2 cd = ptab[idx];           // ds_load_b64 gather
        afrag[j] = (_Float16)fmaf(t, cd.y, cd.x);
      }
      // D = A x B + C  (f32 accumulate)
      acc = __builtin_amdgcn_wmma_f32_16x16x32_f16(
          /*neg_a=*/false, afrag, /*neg_b=*/false, bfr[ch],
          /*c_mod=*/(short)0, acc, /*reuse_a=*/false, /*reuse_b=*/false);
    }

    // Column N==0 lives in lanes 0 (M=0..7) and 16 (M=8..15).
    if (m == 0) {
      const int qrow = qbase + (g ? 8 : 0);
      #pragma unroll
      for (int r = 0; r < 8; ++r) {
        const float s   = acc[r];
        const float u   = s * 63.0f;           // Phi knots: [0,1], h = 1/63
        float       fi  = floorf(u);
        fi              = fminf(fmaxf(fi, 0.0f), 62.0f);
        const int   idx = (int)fi;
        const float t   = u - fi;
        const float2 cd = Ptab[idx];
        out[b * Q_SZ + qrow + r] = fmaf(t, cd.y, cd.x);
      }
    }
  }
}

// ---------------------------------------------------------------------------
extern "C" void kernel_launch(void* const* d_in, const int* in_sizes, int n_in,
                              void* d_out, int out_size, void* d_ws, size_t ws_size,
                              hipStream_t stream) {
  (void)in_sizes; (void)n_in; (void)out_size; (void)ws_size;
  const float* x           = (const float*)d_in[0];
  const float* phi_log_inc = (const float*)d_in[1];
  const float* lambdas     = (const float*)d_in[2];
  const float* Phi_coeffs  = (const float*)d_in[3];
  // d_in[4], d_in[5] (knots) are uniform linspaces; folded into constants.
  float* ws = (float*)d_ws;

  build_tables_kernel<<<1, 32, 0, stream>>>(phi_log_inc, Phi_coeffs, ws);
  sprecher_kernel<<<B_SZ, 256, 0, stream>>>(x, lambdas, ws, (float*)d_out);
}
